// Discriminator_1632087573033
// MI455X (gfx1250) — matrix operand order repaired and checked by reading, not compiled, not run on hardware
//
#include <hip/hip_runtime.h>
#include <hip/hip_bf16.h>
#include <math.h>

// Model dims (fixed by the reference)
#define BB 8
#define LL 512
#define CC 32
#define HH 256
#define DSN 64
#define NLAY 2
#define DII 256
#define RTK 16
#define MROWS (BB * LL)   // 4096

typedef __attribute__((ext_vector_type(16))) _Float16 v16h;
typedef __attribute__((ext_vector_type(8)))  float    v8f;
typedef __attribute__((ext_vector_type(4)))  float    f32x4;
typedef __attribute__((ext_vector_type(4)))  _Float16 f16x4;

// ---------------------------------------------------------------------------
// Tiled WMMA GEMM:  Out[M,N] = epi( A[M,K](lda) * W[N,K]^T + ... )
// 128 threads = 4 waves; block tile 128x64; K staged through LDS in 32-chunks
// as f16 (float4 loads -> packed cvt -> b64 LDS stores). Each wave owns a
// 32-row strip: 2 A-frags + 4 B-frags -> 8 x v_wmma_f32_16x16x32_f16 / chunk.
// EPI: 0 = none, 1 = +bias[n] + pos[(m%L)*N+n], 2 = softplus(x + bias[n])
// GUARD: chunk-granular bounds checks (needs K%4==0); fast path has none.
// ---------------------------------------------------------------------------
#define TM 128
#define TN 64
#define TK 32

template <int EPI, bool GUARD>
__global__ __launch_bounds__(128) void gemm_wmma(
    const float* __restrict__ A, int lda,
    const float* __restrict__ W,            // (N,K) row-major
    const float* __restrict__ bias,         // may be null
    const float* __restrict__ pos,          // may be null
    float* __restrict__ Out,
    int M, int N, int K)
{
    __shared__ _Float16 sA[TM][TK];   // 8 KB
    __shared__ _Float16 sW[TN][TK];   // 4 KB

    const int tid  = threadIdx.x;
    const int wave = tid >> 5;
    const int lane = tid & 31;
    const int hf   = lane >> 4;     // lane half (0/1)
    const int l16  = lane & 15;
    const int row0 = blockIdx.y * TM;
    const int n0   = blockIdx.x * TN;

    v8f acc[2][4] = {};

    for (int k0 = 0; k0 < K; k0 += TK) {
        __syncthreads();
        // ---- Stage A tile (128x32) as f16: 8 float4 per thread.
        {
            f32x4 va[8];
            #pragma unroll
            for (int j = 0; j < 8; ++j) {
                int idx = j * 128 + tid;            // float4 index
                int m  = idx >> 3;                  // /8 chunks per row
                int k4 = (idx & 7) << 2;
                f32x4 v = {};
                if (!GUARD || (k0 + k4 < K))
                    v = *(const f32x4*)&A[(size_t)(row0 + m) * lda + (k0 + k4)];
                va[j] = v;
            }
            #pragma unroll
            for (int j = 0; j < 8; ++j) {
                int idx = j * 128 + tid;
                int m  = idx >> 3;
                int k4 = (idx & 7) << 2;
                f16x4 h;
                #pragma unroll
                for (int e = 0; e < 4; ++e) h[e] = (_Float16)va[j][e];
                *(f16x4*)&sA[m][k4] = h;
            }
        }
        // ---- Stage W tile (64 rows x 32) as f16: 4 float4 per thread.
        {
            f32x4 vw[4];
            #pragma unroll
            for (int j = 0; j < 4; ++j) {
                int idx = j * 128 + tid;
                int n  = idx >> 3;
                int k4 = (idx & 7) << 2;
                f32x4 v = {};
                if (!GUARD || ((n0 + n < N) && (k0 + k4 < K)))
                    v = *(const f32x4*)&W[(size_t)(n0 + n) * K + (k0 + k4)];
                vw[j] = v;
            }
            #pragma unroll
            for (int j = 0; j < 4; ++j) {
                int idx = j * 128 + tid;
                int n  = idx >> 3;
                int k4 = (idx & 7) << 2;
                f16x4 h;
                #pragma unroll
                for (int e = 0; e < 4; ++e) h[e] = (_Float16)vw[j][e];
                *(f16x4*)&sW[n][k4] = h;
            }
        }
        // Prefetch next K-chunk (gfx1250 global_prefetch_b8, speculative).
        if (k0 + TK < K) {
            __builtin_prefetch(&A[(size_t)(row0 + tid) * lda + (k0 + TK)], 0, 1);
            if (!GUARD || (n0 + (tid & 63) < N))
                __builtin_prefetch(&W[(size_t)(n0 + (tid & 63)) * K + (k0 + TK)], 0, 1);
        }
        __syncthreads();

        // A fragments (ISA 7.12.2, 16-bit A 16x32): lane half h, row m=l16,
        // elems 0..7 -> K = h*8+e ; elems 8..15 -> K = 16 + h*8 + (e-8).
        v16h af[2];
        #pragma unroll
        for (int s = 0; s < 2; ++s) {
            const int am = wave * 32 + s * 16 + l16;
            #pragma unroll
            for (int e = 0; e < 8; ++e) {
                af[s][e]     = sA[am][hf * 8 + e];
                af[s][8 + e] = sA[am][16 + hf * 8 + e];
            }
        }
        #pragma unroll
        for (int t = 0; t < 4; ++t) {
            // B fragment: lane half h, elems 0..7 -> K = h*8+e ; 8..15 -> K = 16 + h*8 + (e-8), column n = l16.
            const int bn = t * 16 + l16;
            v16h bf;
            #pragma unroll
            for (int e = 0; e < 16; ++e)
                bf[e] = sW[bn][(e >> 3) * 16 + hf * 8 + (e & 7)];
            #pragma unroll
            for (int s = 0; s < 2; ++s)
                acc[s][t] = __builtin_amdgcn_wmma_f32_16x16x32_f16(
                    false, af[s], false, bf, (short)0, acc[s][t], false, false);
        }
    }

    // Epilogue. C/D layout: VGPR r -> M = hf*8 + r, N = l16.
    #pragma unroll
    for (int s = 0; s < 2; ++s) {
        #pragma unroll
        for (int t = 0; t < 4; ++t) {
            int n = n0 + t * 16 + l16;
            if (GUARD && n >= N) continue;
            #pragma unroll
            for (int r = 0; r < 8; ++r) {
                int m = row0 + wave * 32 + s * 16 + hf * 8 + r;
                float v = acc[s][t][r];
                if (EPI == 1) v += bias[n] + pos[(size_t)(m % LL) * N + n];
                if (EPI == 2) {
                    v += bias[n];
                    v = fmaxf(v, 0.0f) + log1pf(expf(-fabsf(v)));   // softplus
                }
                Out[(size_t)m * N + n] = v;
            }
        }
    }
}

// ---------------------------------------------------------------------------
// LayerNorm over H=256 per row; one 256-thread block per row.
// ---------------------------------------------------------------------------
__global__ __launch_bounds__(256) void layernorm_k(
    const float* __restrict__ x, const float* __restrict__ w,
    const float* __restrict__ b, float* __restrict__ out)
{
    __shared__ float red[256];
    const int row = blockIdx.x;
    const int h   = threadIdx.x;
    float v = x[(size_t)row * HH + h];

    red[h] = v; __syncthreads();
    for (int s = 128; s > 0; s >>= 1) {
        if (h < s) red[h] += red[h + s];
        __syncthreads();
    }
    float mu = red[0] * (1.0f / HH);
    __syncthreads();
    float d = v - mu;
    red[h] = d * d; __syncthreads();
    for (int s = 128; s > 0; s >>= 1) {
        if (h < s) red[h] += red[h + s];
        __syncthreads();
    }
    float var = red[0] * (1.0f / HH);
    out[(size_t)row * HH + h] = d * rsqrtf(var + 1e-5f) * w[h] + b[h];
}

// ---------------------------------------------------------------------------
// Causal depthwise conv (K=2) + bias + SiLU.  Reads xc half of xz (B,L,2*DI).
// ---------------------------------------------------------------------------
__global__ __launch_bounds__(256) void conv_silu_k(
    const float* __restrict__ xz, const float* __restrict__ cw,
    const float* __restrict__ cb, float* __restrict__ out)
{
    const int idx = blockIdx.x * 256 + threadIdx.x;   // over B*L*DI
    const int d  = idx & 255;
    const int bl = idx >> 8;
    const int l  = bl & (LL - 1);
    float cur  = xz[(size_t)bl * (2 * DII) + d];
    float prev = (l > 0) ? xz[(size_t)(bl - 1) * (2 * DII) + d] : 0.0f;
    float v = cw[d * 2 + 0] * prev + cw[d * 2 + 1] * cur + cb[d];
    out[idx] = v * (1.0f / (1.0f + __expf(-v)));      // SiLU
}

// ---------------------------------------------------------------------------
// Selective scan, fused with +u*D and *silu(z).
// One block per batch b, one thread per channel d; h[DS] in VGPRs.
// Per timestep the 128 contiguous B|C floats (x_dbl cols 16..143) are pulled
// into LDS with gfx1250 async global->LDS copies (ASYNCcnt).
// ---------------------------------------------------------------------------
__global__ __launch_bounds__(256) void scan_k(
    const float* __restrict__ u,      // (B,L,DI) conv output
    const float* __restrict__ delta,  // (B,L,DI)
    const float* __restrict__ xdbl,   // (B,L,144): [dt16 | B64 | C64]
    const float* __restrict__ xz,     // (B,L,2*DI) -> z = cols 256..511
    const float* __restrict__ A_log,  // (DI,DS) for this layer
    const float* __restrict__ Dp,     // (DI,)
    float* __restrict__ out)          // (B,L,DI)
{
    __shared__ float sBC[2 * DSN];    // [B64 | C64]
    const int b = blockIdx.x;
    const int d = threadIdx.x;

    float Arow[DSN];
    #pragma unroll
    for (int n = 0; n < DSN; ++n) Arow[n] = -__expf(A_log[d * DSN + n]);
    const float Dd = Dp[d];

    float h[DSN];
    #pragma unroll
    for (int n = 0; n < DSN; ++n) h[n] = 0.0f;

    const unsigned lds_base = (unsigned)(uintptr_t)&sBC[0];

    for (int l = 0; l < LL; ++l) {
        const size_t rb = (size_t)b * LL + l;
        if (d < 2 * DSN) {
            const float* gsrc = &xdbl[rb * 144 + RTK + d];
            unsigned lds_addr = lds_base + 4u * (unsigned)d;
            asm volatile("global_load_async_to_lds_b32 %0, %1, off"
                         :: "v"(lds_addr), "v"(gsrc) : "memory");
        }
        asm volatile("s_wait_asynccnt 0x0" ::: "memory");
        __syncthreads();

        const float dt = delta[rb * DII + d];
        const float uv = u[rb * DII + d];
        const float du = dt * uv;
        float y = 0.0f;
        #pragma unroll
        for (int n = 0; n < DSN; ++n) {
            h[n] = __expf(dt * Arow[n]) * h[n] + du * sBC[n];
            y += h[n] * sBC[DSN + n];
        }
        const float z  = xz[rb * (2 * DII) + DII + d];
        const float sz = z * (1.0f / (1.0f + __expf(-z)));
        out[rb * DII + d] = (y + uv * Dd) * sz;
        __syncthreads();
    }
}

// ---------------------------------------------------------------------------
// Final FC: out[b] = sigmoid( dot(x[b,:], fc_w) + fc_b ), dot over L*H=131072.
// ---------------------------------------------------------------------------
__global__ __launch_bounds__(256) void fc_k(
    const float* __restrict__ x, const float* __restrict__ fw,
    const float* __restrict__ fb, float* __restrict__ out)
{
    __shared__ float red[256];
    const int b = blockIdx.x;
    float s = 0.0f;
    for (int j = threadIdx.x; j < LL * HH; j += 256)
        s += x[(size_t)b * (LL * HH) + j] * fw[j];
    red[threadIdx.x] = s; __syncthreads();
    for (int st = 128; st > 0; st >>= 1) {
        if (threadIdx.x < st) red[threadIdx.x] += red[threadIdx.x + st];
        __syncthreads();
    }
    if (threadIdx.x == 0)
        out[b] = 1.0f / (1.0f + expf(-(red[0] + fb[0])));
}

// ---------------------------------------------------------------------------
extern "C" void kernel_launch(void* const* d_in, const int* in_sizes, int n_in,
                              void* d_out, int out_size, void* d_ws, size_t ws_size,
                              hipStream_t stream)
{
    (void)in_sizes; (void)n_in; (void)out_size; (void)ws_size;

    const float* input    = (const float*)d_in[0];   // (B,L,C)
    const float* l1_w     = (const float*)d_in[1];   // (H,C)
    const float* l1_b     = (const float*)d_in[2];   // (H,)
    const float* pos      = (const float*)d_in[3];   // (1,L,H)
    const float* ln_w     = (const float*)d_in[4];   // (NL,H)
    const float* ln_b     = (const float*)d_in[5];   // (NL,H)
    const float* in_w     = (const float*)d_in[6];   // (NL,2*DI,H)
    const float* conv_w   = (const float*)d_in[7];   // (NL,DI,K)
    const float* conv_b   = (const float*)d_in[8];   // (NL,DI)
    const float* xp_w     = (const float*)d_in[9];   // (NL,RT+2*DS,DI)
    const float* dt_w     = (const float*)d_in[10];  // (NL,DI,RT)
    const float* dt_b     = (const float*)d_in[11];  // (NL,DI)
    const float* A_log    = (const float*)d_in[12];  // (NL,DI,DS)
    const float* Dvec     = (const float*)d_in[13];  // (NL,DI)
    const float* out_w    = (const float*)d_in[14];  // (NL,H,DI)
    const float* fc_w     = (const float*)d_in[15];  // (1,L*H)
    const float* fc_b     = (const float*)d_in[16];  // (1,)

    char* ws = (char*)d_ws;
    float* x     = (float*)(ws + ((size_t) 0 << 20));  // 4 MB  (B,L,H)
    float* xn    = (float*)(ws + ((size_t) 4 << 20));  // 4 MB  LN out
    float* xz    = (float*)(ws + ((size_t) 8 << 20));  // 8 MB  (B,L,2*DI)
    float* xc    = (float*)(ws + ((size_t)16 << 20));  // 4 MB  conv out
    float* xdbl  = (float*)(ws + ((size_t)20 << 20));  // 2.4 MB (B,L,144)
    float* delta = (float*)(ws + ((size_t)23 << 20));  // 4 MB  (B,L,DI)
    float* yscan = (float*)(ws + ((size_t)27 << 20));  // 4 MB  (B,L,DI)

    const dim3 gblk(128);
    const int gy = MROWS / TM;   // 32

    // 1) Embed: x = input @ l1_w^T + l1_b + pos_embed   (M=4096,N=256,K=32)
    gemm_wmma<1, false><<<dim3(HH / TN, gy), gblk, 0, stream>>>(
        input, CC, l1_w, l1_b, pos, x, MROWS, HH, CC);

    for (int i = 0; i < NLAY; ++i) {
        // 2) LayerNorm
        layernorm_k<<<MROWS, 256, 0, stream>>>(x, ln_w + i * HH, ln_b + i * HH, xn);

        // 3) in_proj: xz = xn @ in_proj_w^T    (N=512,K=256)
        gemm_wmma<0, false><<<dim3((2 * DII) / TN, gy), gblk, 0, stream>>>(
            xn, HH, in_w + (size_t)i * 2 * DII * HH, nullptr, nullptr,
            xz, MROWS, 2 * DII, HH);

        // 4) causal dwconv + SiLU
        conv_silu_k<<<(MROWS * DII) / 256, 256, 0, stream>>>(
            xz, conv_w + i * DII * 2, conv_b + i * DII, xc);

        // 5) x_proj: xdbl = xc @ x_proj_w^T    (N=144,K=256) — guarded (N)
        gemm_wmma<0, true><<<dim3((144 + TN - 1) / TN, gy), gblk, 0, stream>>>(
            xc, DII, xp_w + (size_t)i * 144 * DII, nullptr, nullptr,
            xdbl, MROWS, 144, DII);

        // 6) dt_proj + softplus (N=256,K=16) — guarded (K)
        gemm_wmma<2, true><<<dim3(DII / TN, gy), gblk, 0, stream>>>(
            xdbl, 144, dt_w + (size_t)i * DII * RTK, dt_b + i * DII, nullptr,
            delta, MROWS, DII, RTK);

        // 7) selective scan fused with u*D and silu(z) gate
        scan_k<<<BB, DII, 0, stream>>>(
            xc, delta, xdbl, xz, A_log + (size_t)i * DII * DSN, Dvec + i * DII, yscan);

        // 8) out_proj: x = yscan @ out_proj_w^T   (N=256,K=256)
        gemm_wmma<0, false><<<dim3(HH / TN, gy), gblk, 0, stream>>>(
            yscan, DII, out_w + (size_t)i * HH * DII, nullptr, nullptr,
            x, MROWS, HH, DII);
    }

    // 9) FC + sigmoid
    fc_k<<<BB, 256, 0, stream>>>(x, fc_w, fc_b, (float*)d_out);
}
